// DynamisCropClassifier_21406117003575
// MI455X (gfx1250) — matrix operand order repaired and checked by reading, not compiled, not run on hardware
//
#include <hip/hip_runtime.h>
#include <hip/hip_bf16.h>
#include <math.h>

typedef __attribute__((ext_vector_type(16))) _Float16 v16h;
typedef __attribute__((ext_vector_type(8)))  _Float16 v8h;
typedef __attribute__((ext_vector_type(8)))  float    v8f;

#define S7 7
#define HID 64
#define HEADS 4
#define FIN 17
#define NCROPS 3
#define BB 1024
#define TT 128
#define ZK 160            /* 135 padded to 5 K-chunks of 32 */
#define NOISE 0.5f
#define PINIT 0.1f

/* output offsets (floats), concat order: crop, pheno, innov, unc, xfin, traj */
#define OFF_CROP  0
#define OFF_PHENO (BB*NCROPS)
#define OFF_INNOV (OFF_PHENO + BB*TT*S7)
#define OFF_UNC   (OFF_INNOV + BB*TT*S7)
#define OFF_XFIN  (OFF_UNC + BB)
#define OFF_TRAJ  (OFF_XFIN + BB*S7)

/* A-fragment K index for 16-bit 16x32 A: VGPR-pair i, lane-half h */
#define KPB_A(i, h) ((((i) >> 2) * 16) + ((h) * 8) + (((i) & 3) << 1))
/* B-fragment K index for 16-bit 32x16 B: same K order as A */
#define KPB_B(i, h) ((((i) >> 2) * 16) + ((h) * 8) + (((i) & 3) << 1))

#define WMMA_F16(A, Bf, C) \
  __builtin_amdgcn_wmma_f32_16x16x32_f16(false, (A), false, (Bf), (short)0, (C), false, false)

/* ------------------------------------------------------------------ */
/* Kernel 1: h16 = f16(x @ W_in + b_in)  (B*T rows, K=17 pad 32, N=64) */
/* ------------------------------------------------------------------ */
__global__ void k_proj(const float* __restrict__ x, const float* __restrict__ Win,
                       const float* __restrict__ bin, _Float16* __restrict__ h16) {
  const int wave = threadIdx.x >> 5, lane = threadIdx.x & 31;
  const int half = lane >> 4, l16 = lane & 15;
  const int rowBase = blockIdx.x * 64 + wave * 16;
  const float* xr = x + (rowBase + l16) * FIN;
  v16h a;
#pragma unroll
  for (int i = 0; i < 8; ++i) {
    int kb = KPB_A(i, half);
    /* clamp index, select value: branch-free */
    float f0 = xr[kb < FIN ? kb : 0];
    float f1 = xr[kb + 1 < FIN ? kb + 1 : 0];
    a[2*i]   = (_Float16)(kb     < FIN ? f0 : 0.f);
    a[2*i+1] = (_Float16)(kb + 1 < FIN ? f1 : 0.f);
  }
#pragma unroll
  for (int nt = 0; nt < 4; ++nt) {
    const int n = nt * 16 + l16;
    v16h bf;
#pragma unroll
    for (int i = 0; i < 8; ++i) {
      int kb = KPB_B(i, half);
      float f0 = Win[(kb < FIN ? kb : 0) * HID + n];
      float f1 = Win[(kb + 1 < FIN ? kb + 1 : 0) * HID + n];
      bf[2*i]   = (_Float16)(kb     < FIN ? f0 : 0.f);
      bf[2*i+1] = (_Float16)(kb + 1 < FIN ? f1 : 0.f);
    }
    v8f c = {};
    c = WMMA_F16(a, bf, c);
#pragma unroll
    for (int r = 0; r < 8; ++r)
      h16[(rowBase + r + 8 * half) * HID + n] = (_Float16)(c[r] + bin[n]);
  }
}

/* ------------------------------------------------------------------ */
/* Kernel 2: recurrent executor + Kalman scan over T=128               */
/* block = 128 threads (4 waves) handles 16 batch rows                 */
/* h_t tile streamed global->LDS with async-to-LDS (ASYNCcnt)          */
/* ------------------------------------------------------------------ */
__global__ void k_recur(const _Float16* __restrict__ h16,
                        const float* __restrict__ W_exec, const float* __restrict__ b_exec,
                        const float* __restrict__ W_meas, const float* __restrict__ b_meas,
                        const float* __restrict__ A,
                        float* __restrict__ o_innov, float* __restrict__ o_traj,
                        float* __restrict__ o_unc, float* __restrict__ o_xfin) {
  __shared__ __align__(16) _Float16 s_z[16 * ZK];  /* [h_t | x_pred | h_exec | pad] */
  __shared__ float s_hnew[16 * HID];
  __shared__ float s_P[16][49];
  __shared__ float s_Pp[16][49];
  __shared__ float s_K[16][49];
  __shared__ float s_GJ[16][98];           /* predict tmp / Gauss-Jordan aug */
  __shared__ float s_x[16][8];
  __shared__ float s_xp[16][8];
  __shared__ float s_meas[16][8];
  __shared__ float s_A[49];
  __shared__ float s_Wm[HID * S7];
  __shared__ float s_bm[8];
  __shared__ float s_be[HID];

  const int tid = threadIdx.x;
  const int wave = tid >> 5, lane = tid & 31, half = lane >> 4, l16 = lane & 15;
  const int b0 = blockIdx.x * 16;
  const int n = wave * 16 + l16;

  for (int e = tid; e < 49; e += 128) s_A[e] = A[e];
  for (int e = tid; e < HID * S7; e += 128) s_Wm[e] = W_meas[e];
  if (tid < S7) s_bm[tid] = b_meas[tid];
  for (int e = tid; e < HID; e += 128) s_be[e] = b_exec[e];
  for (int e = tid; e < 16 * ZK; e += 128)
    if ((e % ZK) >= HID) s_z[e] = (_Float16)0.f;      /* zero x_pred/h_exec/pad */
  if (tid < 16) {
#pragma unroll
    for (int i = 0; i < S7; ++i) {
      s_x[tid][i] = 0.f;
#pragma unroll
      for (int j = 0; j < S7; ++j) s_P[tid][i * 7 + j] = (i == j) ? PINIT : 0.f;
    }
  }
  /* W_exec B-fragments: resident in registers for all 128 steps */
  v16h bw[5];
#pragma unroll
  for (int c = 0; c < 5; ++c)
#pragma unroll
    for (int i = 0; i < 8; ++i) {
      int kb = c * 32 + KPB_B(i, half);
      float w0 = W_exec[(kb < 135 ? kb : 0) * HID + n];
      float w1 = W_exec[(kb + 1 < 135 ? kb + 1 : 0) * HID + n];
      bw[c][2*i]   = (_Float16)(kb     < 135 ? w0 : 0.f);
      bw[c][2*i+1] = (_Float16)(kb + 1 < 135 ? w1 : 0.f);
    }
  /* async copy setup: thread -> one 16B chunk of the 16x(64 f16) tile */
  const unsigned arow = (unsigned)tid >> 3, achk = (unsigned)tid & 7;
  const unsigned ldsoff = (unsigned)(uintptr_t)(&s_z[0]) + arow * (ZK * 2) + achk * 16;
  const unsigned gbase = ((unsigned)(b0 + (int)arow) * TT) * (HID * 2) + achk * 16;
  __syncthreads();

  for (int t = 0; t < TT; ++t) {
    /* (2a) kick off async global->LDS copy of h_t tile (one b128/lane) */
    {
      unsigned voff = gbase + (unsigned)t * (HID * 2);
      asm volatile("global_load_async_to_lds_b128 %0, %1, %2"
                   :: "v"(ldsoff), "v"(voff), "s"(h16) : "memory");
    }
    /* (1) Kalman predict, lane-per-batch-row (overlaps the async copy) */
    if (tid < 16) {
      const int row = tid;
#pragma unroll
      for (int i = 0; i < 7; ++i) {
        float s = 0.f;
#pragma unroll
        for (int j = 0; j < 7; ++j) s += s_A[i * 7 + j] * s_x[row][j];
        s_xp[row][i] = s;
        s_z[row * ZK + HID + i] = (_Float16)s;
      }
#pragma unroll
      for (int i = 0; i < 7; ++i)
#pragma unroll
        for (int k = 0; k < 7; ++k) {
          float s = 0.f;
#pragma unroll
          for (int j = 0; j < 7; ++j) s += s_A[i * 7 + j] * s_P[row][j * 7 + k];
          s_GJ[row][i * 7 + k] = s;
        }
#pragma unroll
      for (int i = 0; i < 7; ++i)
#pragma unroll
        for (int l = 0; l < 7; ++l) {
          float s = (i == l) ? NOISE : 0.f;
#pragma unroll
          for (int k = 0; k < 7; ++k) s += s_GJ[row][i * 7 + k] * s_A[l * 7 + k];
          s_Pp[row][i * 7 + l] = s;
        }
    }
    /* (2b) wait for this wave's async tile, then block barrier */
    asm volatile("s_wait_asynccnt 0x0" ::: "memory");
    __syncthreads();
    /* (3) A-fragments from z */
    v16h af[5];
#pragma unroll
    for (int c = 0; c < 5; ++c)
#pragma unroll
      for (int i = 0; i < 8; ++i) {
        int kb = c * 32 + KPB_A(i, half);
        af[c][2*i]   = s_z[l16 * ZK + kb];
        af[c][2*i+1] = s_z[l16 * ZK + kb + 1];
      }
    __syncthreads();                       /* all z reads done before writes */
    /* (4) executor GEMM + tanh */
    v8f acc = {};
#pragma unroll
    for (int c = 0; c < 5; ++c) acc = WMMA_F16(af[c], bw[c], acc);
#pragma unroll
    for (int r = 0; r < 8; ++r) {
      int m = r + 8 * half;
      float hn = tanhf(acc[r] + s_be[n]);
      s_hnew[m * HID + n] = hn;
      s_z[m * ZK + 71 + n] = (_Float16)hn;     /* h_exec slice for next step */
    }
    __syncthreads();
    /* (5) meas = h_new @ W_meas + b_meas */
    for (int e = tid; e < 16 * S7; e += 128) {
      int row = e / 7, s = e % 7;
      float a2 = s_bm[s];
      for (int k = 0; k < HID; ++k) a2 += s_hnew[row * HID + k] * s_Wm[k * S7 + s];
      s_meas[row][s] = a2;
    }
    __syncthreads();
    /* (6) Kalman update */
    if (tid < 16) {
      const int row = tid;
      float innov[7];
#pragma unroll
      for (int s = 0; s < 7; ++s) innov[s] = s_meas[row][s] - s_xp[row][s];
      float* W = s_GJ[row];                /* augmented [S | I] 7x14 */
#pragma unroll
      for (int i = 0; i < 7; ++i)
#pragma unroll
        for (int j = 0; j < 7; ++j) {
          W[i * 14 + j] = s_Pp[row][i * 7 + j] + ((i == j) ? NOISE : 0.f);
          W[i * 14 + 7 + j] = (i == j) ? 1.f : 0.f;
        }
#pragma unroll
      for (int c = 0; c < 7; ++c) {
        float inv = 1.f / W[c * 14 + c];
#pragma unroll
        for (int j = 0; j < 14; ++j) W[c * 14 + j] *= inv;
#pragma unroll
        for (int r2 = 0; r2 < 7; ++r2) {
          if (r2 == c) continue;
          float f = W[r2 * 14 + c];
#pragma unroll
          for (int j = 0; j < 14; ++j) W[r2 * 14 + j] -= f * W[c * 14 + j];
        }
      }
      /* K = P_pred @ S^{-1} */
#pragma unroll
      for (int i = 0; i < 7; ++i)
#pragma unroll
        for (int j = 0; j < 7; ++j) {
          float s = 0.f;
#pragma unroll
          for (int k = 0; k < 7; ++k) s += s_Pp[row][i * 7 + k] * W[k * 14 + 7 + j];
          s_K[row][i * 7 + j] = s;
        }
#pragma unroll
      for (int i = 0; i < 7; ++i) {
        float s = s_xp[row][i];
#pragma unroll
        for (int j = 0; j < 7; ++j) s += s_K[row][i * 7 + j] * innov[j];
        s_x[row][i] = s;
        o_traj[((b0 + row) * TT + t) * S7 + i] = s;
        o_innov[((b0 + row) * TT + t) * S7 + i] = innov[i];
      }
#pragma unroll
      for (int i = 0; i < 7; ++i)
#pragma unroll
        for (int l = 0; l < 7; ++l) {
          float s = s_Pp[row][i * 7 + l];
#pragma unroll
          for (int k = 0; k < 7; ++k) s -= s_K[row][i * 7 + k] * s_Pp[row][k * 7 + l];
          s_P[row][i * 7 + l] = s;
        }
    }
    __syncthreads();
  }
  if (tid < 16) {
    float tr = 0.f;
#pragma unroll
    for (int i = 0; i < 7; ++i) {
      tr += s_P[tid][i * 7 + i];
      o_xfin[(b0 + tid) * S7 + i] = s_x[tid][i];
    }
    o_unc[b0 + tid] = tr;
  }
}

/* ------------------------------------------------------------------ */
/* Kernel 3: gated attention + heads; one block per batch row          */
/* block = 64 threads (2 waves)                                        */
/* ------------------------------------------------------------------ */
__global__ void k_attn(const _Float16* __restrict__ h16, const unsigned char* __restrict__ mask,
                       const float* __restrict__ hurst,
                       const float* __restrict__ W_attn, const float* __restrict__ b_attn,
                       const float* __restrict__ Wq, const float* __restrict__ bq,
                       const float* __restrict__ Wk, const float* __restrict__ bk,
                       const float* __restrict__ Wv, const float* __restrict__ bv,
                       const float* __restrict__ Wo, const float* __restrict__ bo,
                       const float* __restrict__ W_phys, const float* __restrict__ b_phys,
                       const float* __restrict__ ln_g, const float* __restrict__ ln_b,
                       const float* __restrict__ W_crop, const float* __restrict__ b_crop,
                       const float* __restrict__ W_pheno, const float* __restrict__ b_pheno,
                       const float* __restrict__ traj, const float* __restrict__ innov,
                       float* __restrict__ o_crop, float* __restrict__ o_pheno) {
  __shared__ _Float16 s_q[TT * HID];   /* a_in -> q -> ctx (in place) */
  __shared__ _Float16 s_k[TT * HID];
  __shared__ _Float16 s_v[TT * HID];
  __shared__ __align__(16) float s_scr[2][1024];  /* per-wave: a_in staging / probs / attn_out */
  __shared__ float s_pool[HID];
  __shared__ float s_red[64];
  __shared__ float s_gate[HEADS];
  __shared__ float s_stat[2];          /* chaos, cnt */

  const int tid = threadIdx.x;
  const int wave = tid >> 5, lane = tid & 31, half = lane >> 4, l16 = lane & 15;
  const int b = blockIdx.x;

  s_pool[tid] = 0.f;
  {
    float p = 0.f;
    for (int e = tid; e < TT * S7; e += 64) {
      float v = innov[b * TT * S7 + e];
      p += v * v;
    }
    s_red[tid] = p;
    __syncthreads();
    if (tid == 0) {
      float s = 0.f;
      for (int i = 0; i < 64; ++i) s += s_red[i];
      float mean = s / (float)(TT * S7);
      s_stat[0] = fminf(fmaxf(mean, 0.f), 10.f) / 10.f;
      float cnt = 0.f;
      for (int t2 = 0; t2 < TT; ++t2) cnt += mask[b * TT + t2] ? 1.f : 0.f;
      s_stat[1] = fmaxf(cnt, 1e-6f);
    }
    __syncthreads();
    if (tid < HEADS) {
      float z0 = s_stat[0] * W_phys[tid] + hurst[b] * W_phys[HEADS + tid] + b_phys[tid];
      s_gate[tid] = 1.f / (1.f + expf(-z0));
    }
  }

  /* ---- a_in, then k, v, q (q overwrites a_in rows in place) ---- */
  _Float16* stg = (_Float16*)s_scr[wave];   /* 16 x 96 f16 concat staging */
  for (int e = lane; e < 16 * 25; e += 32) { /* zero pad cols 71..95 once */
    int row = e / 25, col = 71 + e % 25;
    stg[row * 96 + col] = (_Float16)0.f;
  }
  for (int mt = wave; mt < 8; mt += 2) {
    const int t0 = mt * 16;
    const int trow = t0 + l16;
    /* stage concat [h16 | f16(traj)] into LDS: vectorized + branch-free */
    for (int e = lane; e < 16 * 8; e += 32) {
      int row = e >> 3, chk = e & 7;
      *(v8h*)&stg[row * 96 + chk * 8] =
          *(const v8h*)&h16[(b * TT + t0 + row) * HID + chk * 8];
    }
    for (int e = lane; e < 16 * S7; e += 32) {
      int row = e / 7, s = e % 7;
      stg[row * 96 + HID + s] = (_Float16)traj[(b * TT + t0 + row) * S7 + s];
    }
    v16h aA[3];
#pragma unroll
    for (int c = 0; c < 3; ++c)
#pragma unroll
      for (int i = 0; i < 8; ++i) {
        int kb = c * 32 + KPB_A(i, half);
        aA[c][2*i]   = stg[l16 * 96 + kb];
        aA[c][2*i+1] = stg[l16 * 96 + kb + 1];
      }
#pragma unroll
    for (int nt = 0; nt < 4; ++nt) {
      const int nn = nt * 16 + l16;
      v8f acc = {};
#pragma unroll
      for (int c = 0; c < 3; ++c) {
        v16h bf;
#pragma unroll
        for (int i = 0; i < 8; ++i) {
          int kb = c * 32 + KPB_B(i, half);
          float f0 = W_attn[(kb < 71 ? kb : 0) * HID + nn];
          float f1 = W_attn[(kb + 1 < 71 ? kb + 1 : 0) * HID + nn];
          bf[2*i]   = (_Float16)(kb     < 71 ? f0 : 0.f);
          bf[2*i+1] = (_Float16)(kb + 1 < 71 ? f1 : 0.f);
        }
        acc = WMMA_F16(aA[c], bf, acc);
      }
#pragma unroll
      for (int r = 0; r < 8; ++r)
        s_q[(t0 + r + 8 * half) * HID + nn] = (_Float16)(acc[r] + b_attn[nn]);
    }
    /* a_in fragments (row-local) */
    v16h ain[2];
#pragma unroll
    for (int c = 0; c < 2; ++c)
#pragma unroll
      for (int i = 0; i < 8; ++i) {
        int kb = c * 32 + KPB_A(i, half);
        ain[c][2*i]   = s_q[trow * HID + kb];
        ain[c][2*i+1] = s_q[trow * HID + kb + 1];
      }
    const float* Wt[3] = {Wk, Wv, Wq};
    const float* bt[3] = {bk, bv, bq};
    _Float16* dst[3] = {s_k, s_v, s_q};
#pragma unroll
    for (int p = 0; p < 3; ++p) {
#pragma unroll
      for (int nt = 0; nt < 4; ++nt) {
        const int nn = nt * 16 + l16;
        v8f acc = {};
#pragma unroll
        for (int c = 0; c < 2; ++c) {
          v16h bf;
#pragma unroll
          for (int i = 0; i < 8; ++i) {
            int kb = c * 32 + KPB_B(i, half);   /* < 64 always */
            bf[2*i]   = (_Float16)Wt[p][kb * HID + nn];
            bf[2*i+1] = (_Float16)Wt[p][(kb + 1) * HID + nn];
          }
          acc = WMMA_F16(ain[c], bf, acc);
        }
#pragma unroll
        for (int r = 0; r < 8; ++r)
          dst[p][(t0 + r + 8 * half) * HID + nn] = (_Float16)(acc[r] + bt[p][nn]);
      }
    }
  }
  __syncthreads();

  /* ---- attention + output heads ---- */
  const float invcnt = 1.f / s_stat[1];
  float* scr = s_scr[wave];
  _Float16* pr = (_Float16*)scr;
  for (int mt = wave; mt < 8; mt += 2) {
    const int t0 = mt * 16;
#pragma unroll 1
    for (int hh = 0; hh < HEADS; ++hh) {
      /* q A-frag: K = head-dim 16 padded to 32 (upper half zero, uniform) */
      v16h qa;
#pragma unroll
      for (int i = 0; i < 8; ++i) {
        int kb = KPB_A(i, half);
        _Float16 q0 = s_q[(t0 + l16) * HID + hh * 16 + (kb & 15)];
        _Float16 q1 = s_q[(t0 + l16) * HID + hh * 16 + ((kb + 1) & 15)];
        qa[2*i]   = (kb     < 16) ? q0 : (_Float16)0.f;
        qa[2*i+1] = (kb + 1 < 16) ? q1 : (_Float16)0.f;
      }
      v8f lt[8];
#pragma unroll
      for (int nt = 0; nt < 8; ++nt) {
        const int kp = nt * 16 + l16;
        v16h kf;
#pragma unroll
        for (int i = 0; i < 8; ++i) {
          int kk = KPB_B(i, half);
          _Float16 k0 = s_k[kp * HID + hh * 16 + (kk & 15)];
          _Float16 k1 = s_k[kp * HID + hh * 16 + ((kk + 1) & 15)];
          kf[2*i]   = (kk     < 16) ? k0 : (_Float16)0.f;
          kf[2*i+1] = (kk + 1 < 16) ? k1 : (_Float16)0.f;
        }
        v8f z = {};
        lt[nt] = WMMA_F16(qa, kf, z);
      }
      const float g = s_gate[hh] * 0.25f;  /* gate * 1/sqrt(hd) */
      float mk[8];
#pragma unroll
      for (int nt = 0; nt < 8; ++nt) mk[nt] = mask[b * TT + nt * 16 + l16] ? 1.f : 0.f;
#pragma unroll
      for (int r = 0; r < 8; ++r) {
        float mx = -3.0e38f;
#pragma unroll
        for (int nt = 0; nt < 8; ++nt) {
          float val = (mk[nt] > 0.f) ? lt[nt][r] * g : -1e9f;
          lt[nt][r] = val;
          mx = fmaxf(mx, val);
        }
        mx = fmaxf(mx, __shfl_xor(mx, 1, 32));
        mx = fmaxf(mx, __shfl_xor(mx, 2, 32));
        mx = fmaxf(mx, __shfl_xor(mx, 4, 32));
        mx = fmaxf(mx, __shfl_xor(mx, 8, 32));
        float sm = 0.f;
#pragma unroll
        for (int nt = 0; nt < 8; ++nt) {
          float e = expf(lt[nt][r] - mx);
          lt[nt][r] = e;
          sm += e;
        }
        sm += __shfl_xor(sm, 1, 32);
        sm += __shfl_xor(sm, 2, 32);
        sm += __shfl_xor(sm, 4, 32);
        sm += __shfl_xor(sm, 8, 32);
        const float inv = 1.f / sm;
#pragma unroll
        for (int nt = 0; nt < 8; ++nt)
          pr[(r + 8 * half) * TT + nt * 16 + l16] = (_Float16)(lt[nt][r] * inv);
      }
      /* ctx = probs @ v_head  (K=128, N=16) */
      v8f cacc = {};
#pragma unroll
      for (int c = 0; c < 4; ++c) {
        v16h pa, vb;
#pragma unroll
        for (int i = 0; i < 8; ++i) {
          int ka = c * 32 + KPB_A(i, half);
          pa[2*i]   = pr[l16 * TT + ka];
          pa[2*i+1] = pr[l16 * TT + ka + 1];
          int kb = c * 32 + KPB_B(i, half);
          vb[2*i]   = s_v[kb * HID + hh * 16 + l16];
          vb[2*i+1] = s_v[(kb + 1) * HID + hh * 16 + l16];
        }
        cacc = WMMA_F16(pa, vb, cacc);
      }
#pragma unroll
      for (int r = 0; r < 8; ++r)
        s_q[(t0 + r + 8 * half) * HID + hh * 16 + l16] = (_Float16)cacc[r];
    }
    /* attn_out = ctx @ Wo + bo */
    v16h ca[2];
#pragma unroll
    for (int c = 0; c < 2; ++c)
#pragma unroll
      for (int i = 0; i < 8; ++i) {
        int kb = c * 32 + KPB_A(i, half);
        ca[c][2*i]   = s_q[(t0 + l16) * HID + kb];
        ca[c][2*i+1] = s_q[(t0 + l16) * HID + kb + 1];
      }
#pragma unroll
    for (int nt = 0; nt < 4; ++nt) {
      const int nn = nt * 16 + l16;
      v8f acc = {};
#pragma unroll
      for (int c = 0; c < 2; ++c) {
        v16h bf;
#pragma unroll
        for (int i = 0; i < 8; ++i) {
          int kb = c * 32 + KPB_B(i, half);   /* < 64 always */
          bf[2*i]   = (_Float16)Wo[kb * HID + nn];
          bf[2*i+1] = (_Float16)Wo[(kb + 1) * HID + nn];
        }
        acc = WMMA_F16(ca[c], bf, acc);
      }
#pragma unroll
      for (int r = 0; r < 8; ++r)
        scr[(r + 8 * half) * HID + nn] = acc[r] + bo[nn];
    }
    /* pheno logits (unmasked) */
    for (int e = lane; e < 16 * S7; e += 32) {
      int row = e / 7, s = e % 7;
      float a2 = b_pheno[s];
      for (int k = 0; k < HID; ++k) a2 += scr[row * HID + k] * W_pheno[k * S7 + s];
      o_pheno[(b * TT + t0 + row) * S7 + s] = a2;
    }
    /* masked pooled sum */
    for (int c2 = lane; c2 < HID; c2 += 32) {
      float s = 0.f;
      for (int row = 0; row < 16; ++row) {
        float mm = mask[b * TT + t0 + row] ? 1.f : 0.f;
        s += mm * scr[row * HID + c2];
      }
      atomicAdd(&s_pool[c2], s);
    }
  }
  __syncthreads();
  /* LN + crop head (wave 0) */
  if (tid < 32) {
    float p0 = s_pool[tid] * invcnt;
    float p1 = s_pool[tid + 32] * invcnt;
    float sm = p0 + p1;
    for (int off = 1; off < 32; off <<= 1) sm += __shfl_xor(sm, off, 32);
    float mu = sm * (1.f / 64.f);
    float d0 = p0 - mu, d1 = p1 - mu;
    float vv = d0 * d0 + d1 * d1;
    for (int off = 1; off < 32; off <<= 1) vv += __shfl_xor(vv, off, 32);
    float rs = rsqrtf(vv * (1.f / 64.f) + 1e-5f);
    s_red[tid]      = d0 * rs * ln_g[tid]      + ln_b[tid];
    s_red[tid + 32] = d1 * rs * ln_g[tid + 32] + ln_b[tid + 32];
    if (tid < NCROPS) {
      float a2 = b_crop[tid];
      for (int c2 = 0; c2 < HID; ++c2) a2 += s_red[c2] * W_crop[c2 * NCROPS + tid];
      o_crop[b * NCROPS + tid] = a2;
    }
  }
}

/* ------------------------------------------------------------------ */
extern "C" void kernel_launch(void* const* d_in, const int* in_sizes, int n_in,
                              void* d_out, int out_size, void* d_ws, size_t ws_size,
                              hipStream_t stream) {
  const float* x       = (const float*)d_in[0];
  const unsigned char* mask = (const unsigned char*)d_in[1];
  const float* hurst   = (const float*)d_in[2];
  const float* W_in    = (const float*)d_in[3];
  const float* b_in    = (const float*)d_in[4];
  const float* W_exec  = (const float*)d_in[5];
  const float* b_exec  = (const float*)d_in[6];
  const float* W_meas  = (const float*)d_in[7];
  const float* b_meas  = (const float*)d_in[8];
  const float* A       = (const float*)d_in[9];
  const float* W_attn  = (const float*)d_in[10];
  const float* b_attn  = (const float*)d_in[11];
  const float* Wq      = (const float*)d_in[12];
  const float* bq      = (const float*)d_in[13];
  const float* Wk      = (const float*)d_in[14];
  const float* bk      = (const float*)d_in[15];
  const float* Wv      = (const float*)d_in[16];
  const float* bv      = (const float*)d_in[17];
  const float* Wo      = (const float*)d_in[18];
  const float* bo      = (const float*)d_in[19];
  const float* W_phys  = (const float*)d_in[20];
  const float* b_phys  = (const float*)d_in[21];
  const float* ln_g    = (const float*)d_in[22];
  const float* ln_b    = (const float*)d_in[23];
  const float* W_crop  = (const float*)d_in[24];
  const float* b_crop  = (const float*)d_in[25];
  const float* W_pheno = (const float*)d_in[26];
  const float* b_pheno = (const float*)d_in[27];
  float* out = (float*)d_out;
  _Float16* h16 = (_Float16*)d_ws;   /* (B*T, 64) f16 staging */

  k_proj<<<BB * TT / 64, 128, 0, stream>>>(x, W_in, b_in, h16);
  k_recur<<<BB / 16, 128, 0, stream>>>(h16, W_exec, b_exec, W_meas, b_meas, A,
                                       out + OFF_INNOV, out + OFF_TRAJ,
                                       out + OFF_UNC, out + OFF_XFIN);
  k_attn<<<BB, 64, 0, stream>>>(h16, mask, hurst, W_attn, b_attn, Wq, bq, Wk, bk,
                                Wv, bv, Wo, bo, W_phys, b_phys, ln_g, ln_b,
                                W_crop, b_crop, W_pheno, b_pheno,
                                out + OFF_TRAJ, out + OFF_INNOV,
                                out + OFF_CROP, out + OFF_PHENO);
}
